// Mamba2Block_39230231282054
// MI455X (gfx1250) — matrix operand order repaired and checked by reading, not compiled, not run on hardware
//
#include <hip/hip_runtime.h>
#include <stdint.h>

#define BATCH 4
#define SEQ 2048
#define D_MODEL 512
#define D_STATE 128
#define D_CONV 4
#define HEADDIM 64
#define D_INNER 1024
#define NHEADS 16
#define CONV_DIM (D_INNER + 2 * D_STATE)               /* 1280 */
#define D_IN_PROJ (2 * D_INNER + 2 * D_STATE + NHEADS) /* 2320 */
#define D_IN_PROJ_PAD 2432                             /* 19 * 128 */
#define ROWS (BATCH * SEQ)                             /* 8192 */

typedef __attribute__((ext_vector_type(16))) __bf16 v16bf;
typedef __attribute__((ext_vector_type(8)))  float  v8f;

struct B32x8 { uint32_t d[8]; };

__device__ __forceinline__ unsigned short f2bf(float f) {
  union { float f; unsigned int u; } v; v.f = f;
  unsigned int u = v.u;
  unsigned int r = u + 0x7FFFu + ((u >> 16) & 1u);   // round-to-nearest-even
  return (unsigned short)(r >> 16);
}

__device__ __forceinline__ float silu(float x) {
  return x * (1.0f / (1.0f + __expf(-x)));
}

__device__ __forceinline__ float block_reduce_sum(float v, float* smem) {
#pragma unroll
  for (int off = 16; off; off >>= 1) v += __shfl_xor(v, off, 32);
  int lane = threadIdx.x & 31, w = threadIdx.x >> 5;
  if (lane == 0) smem[w] = v;
  __syncthreads();
  float s = (threadIdx.x < 8) ? smem[threadIdx.x] : 0.0f;
  if (w == 0) {
#pragma unroll
    for (int off = 4; off; off >>= 1) s += __shfl_xor(s, off, 32);
    if (lane == 0) smem[0] = s;
  }
  __syncthreads();
  return smem[0];
}

// --- CDNA5 async copy global->LDS, tracked by ASYNCcnt -----------------------
__device__ __forceinline__ void async_b128(uint32_t lds_addr, uint64_t gaddr) {
  asm volatile("global_load_async_to_lds_b128 %0, %1, off"
               :: "v"(lds_addr), "v"(gaddr)
               : "memory");
}
#if __has_builtin(__builtin_amdgcn_s_wait_asynccnt)
#define WAIT_ASYNC(n) __builtin_amdgcn_s_wait_asynccnt(n)
#else
#define WAIT_ASYNC(n) asm volatile("s_wait_asynccnt %0" :: "i"(n) : "memory")
#endif

// ------------------------------------------------ f32 -> bf16 with row padding
// out has Npad rows of K; rows >= Nreal are zero-filled.
__global__ void convert_pad_bf16_kernel(const float* __restrict__ in,
                                        uint16_t* __restrict__ out,
                                        int Nreal, int Npad, int K) {
  size_t i = (size_t)blockIdx.x * blockDim.x + threadIdx.x;
  if (i >= (size_t)Npad * K) return;
  int n = (int)(i / K);
  out[i] = (n < Nreal) ? f2bf(in[i]) : (unsigned short)0;
}

// ------------------------------------------------------- RMSNorm(x) -> bf16 u
__global__ void rmsnorm_bf16_kernel(const float* __restrict__ x,
                                    const float* __restrict__ w,
                                    uint16_t* __restrict__ u) {
  int row = blockIdx.x;
  const float* xr = x + (size_t)row * D_MODEL;
  float v0 = xr[threadIdx.x];
  float v1 = xr[threadIdx.x + 256];
  __shared__ float red[8];
  float ss = block_reduce_sum(v0 * v0 + v1 * v1, red);
  float scale = rsqrtf(ss * (1.0f / D_MODEL) + 1e-6f);
  uint16_t* ur = u + (size_t)row * D_MODEL;
  ur[threadIdx.x]       = f2bf(w[threadIdx.x]       * v0 * scale);
  ur[threadIdx.x + 256] = f2bf(w[threadIdx.x + 256] * v1 * scale);
}

// ------------------------------------------------- bf16 WMMA GEMM  C = A*W^T
// A: M x K (row-major bf16), W: Npad x K (row-major bf16), C: M x N (f32).
// 256-thread block computes a 128x128 tile; K sliced in 32-wide steps that are
// double-buffered through LDS via global_load_async_to_lds_b128 (ASYNCcnt).
// Each wave computes a 32x64 sub-tile = 2x4 WMMA tiles per K-step.
#define LDSROW 80                           /* 64B of bf16 + 16B pad, 16B-aligned */
#define TILEBYTES (128 * LDSROW)            /* one operand tile  */
#define BUFBYTES  (2 * TILEBYTES)           /* A tile + B tile   */

__global__ void __launch_bounds__(256)
gemm_bf16_wmma_kernel(const uint16_t* __restrict__ A,
                      const uint16_t* __restrict__ W,
                      float* __restrict__ C,
                      const float* __restrict__ resid,
                      int M, int N, int Npad, int K) {
  __shared__ uint8_t lds[2 * BUFBYTES];     // 40 KB: 2 buffers x (A+B)
  const int nblk = Npad >> 7;
  const int mt = blockIdx.x / nblk;
  const int nt = blockIdx.x - mt * nblk;
  const int m0 = mt << 7, n0 = nt << 7;
  const int t = threadIdx.x;
  const int wave = t >> 5, lane = t & 31;
  const int half = lane >> 4, idx = lane & 15;
  const int wm = wave & 3;                  // 4 M strips of 32 rows
  const int wn = wave >> 2;                 // 2 N strips of 64 cols
  const uint32_t ldsbase = (uint32_t)(size_t)(&lds[0]);

  // per-thread staging chunks: 512 16B-chunks per operand tile, 2 per thread
  const int c0 = 2 * t;
  const int rowS0 = c0 >> 2,     colS0 = (c0 & 3) * 16;
  const int rowS1 = (c0 + 1) >> 2, colS1 = ((c0 + 1) & 3) * 16;
  const uint64_t Ab = (uint64_t)(size_t)(A + (size_t)m0 * K);
  const uint64_t Wb = (uint64_t)(size_t)(W + (size_t)n0 * K);
  const size_t rowBytes = (size_t)K * 2;

  v8f acc[2][4];
#pragma unroll
  for (int mi = 0; mi < 2; ++mi)
#pragma unroll
    for (int ni = 0; ni < 4; ++ni) acc[mi][ni] = (v8f){};

  const int nk = K >> 5;
  // --- stage step 0 into buffer 0
  {
    uint32_t lb = ldsbase;
    async_b128(lb + rowS0 * LDSROW + colS0, Ab + (size_t)rowS0 * rowBytes + colS0);
    async_b128(lb + rowS1 * LDSROW + colS1, Ab + (size_t)rowS1 * rowBytes + colS1);
    async_b128(lb + TILEBYTES + rowS0 * LDSROW + colS0, Wb + (size_t)rowS0 * rowBytes + colS0);
    async_b128(lb + TILEBYTES + rowS1 * LDSROW + colS1, Wb + (size_t)rowS1 * rowBytes + colS1);
  }
  for (int ks = 0; ks < nk; ++ks) {
    if (ks + 1 < nk) {                      // stage next slice into other buffer
      uint32_t lb = ldsbase + ((ks + 1) & 1) * BUFBYTES;
      uint64_t ka = Ab + (size_t)(ks + 1) * 64;   // 32 bf16 = 64 bytes
      uint64_t kb = Wb + (size_t)(ks + 1) * 64;
      async_b128(lb + rowS0 * LDSROW + colS0, ka + (size_t)rowS0 * rowBytes + colS0);
      async_b128(lb + rowS1 * LDSROW + colS1, ka + (size_t)rowS1 * rowBytes + colS1);
      async_b128(lb + TILEBYTES + rowS0 * LDSROW + colS0, kb + (size_t)rowS0 * rowBytes + colS0);
      async_b128(lb + TILEBYTES + rowS1 * LDSROW + colS1, kb + (size_t)rowS1 * rowBytes + colS1);
      WAIT_ASYNC(4);                        // stage ks complete; ks+1 in flight
    } else {
      WAIT_ASYNC(0);
    }
    __syncthreads();                        // publish LDS to all waves

    const uint8_t* tA = &lds[(ks & 1) * BUFBYTES];
    const uint8_t* tB = tA + TILEBYTES;
    B32x8 af[2], bfr[4];
#pragma unroll
    for (int mi = 0; mi < 2; ++mi) {
      int row = wm * 32 + mi * 16 + idx;
#pragma unroll
      for (int i = 0; i < 8; ++i) {
        int ka = ((i < 4) ? 2 * i : 8 + 2 * i) + 8 * half;   // A frag layout
        af[mi].d[i] = *(const uint32_t*)(tA + row * LDSROW + ka * 2);
      }
    }
#pragma unroll
    for (int ni = 0; ni < 4; ++ni) {
      int row = wn * 64 + ni * 16 + idx;
#pragma unroll
      for (int i = 0; i < 8; ++i) {
        int kb = ((i < 4) ? 2 * i : 8 + 2 * i) + 8 * half;   // B frag layout
        bfr[ni].d[i] = *(const uint32_t*)(tB + row * LDSROW + kb * 2);
      }
    }
#pragma unroll
    for (int mi = 0; mi < 2; ++mi)
#pragma unroll
      for (int ni = 0; ni < 4; ++ni)
        acc[mi][ni] = __builtin_amdgcn_wmma_f32_16x16x32_bf16(
            false, __builtin_bit_cast(v16bf, af[mi]),
            false, __builtin_bit_cast(v16bf, bfr[ni]),
            (short)0, acc[mi][ni], false, false);
    __syncthreads();                        // protect buffer before re-stage
  }

  // D layout: lane (half,idx) holds N = idx, rows M = r + 8*half in VGPR r
#pragma unroll
  for (int mi = 0; mi < 2; ++mi)
#pragma unroll
    for (int ni = 0; ni < 4; ++ni)
#pragma unroll
      for (int r = 0; r < 8; ++r) {
        int m = m0 + wm * 32 + mi * 16 + r + 8 * half;
        int n = n0 + wn * 64 + ni * 16 + idx;
        if (n < N) {
          size_t o = (size_t)m * N + n;
          float v = acc[mi][ni][r];
          if (resid) v += resid[o];
          C[o] = v;
        }
      }
}

// ------------------------------------- depthwise causal conv(4) + bias + SiLU
__global__ void conv_silu_kernel(const float* __restrict__ zxbcdt,
                                 const float* __restrict__ cw,
                                 const float* __restrict__ cb,
                                 float* __restrict__ xBC) {
  size_t idx = (size_t)blockIdx.x * blockDim.x + threadIdx.x;
  if (idx >= (size_t)ROWS * CONV_DIM) return;
  int c   = (int)(idx % CONV_DIM);
  int row = (int)(idx / CONV_DIM);
  int l   = row & (SEQ - 1);
  float acc = cb[c];
  const float* w = cw + c * D_CONV;
#pragma unroll
  for (int j = 0; j < D_CONV; ++j) {
    int ls = l - (D_CONV - 1) + j;
    if (ls >= 0)
      acc += w[j] * zxbcdt[(size_t)(row - (D_CONV - 1) + j) * D_IN_PROJ + D_INNER + c];
  }
  xBC[idx] = silu(acc);
}

// ------------------------------------------ dt = softplus(raw+bias); dA = e^{dt*A}
__global__ void dt_kernel(const float* __restrict__ zxbcdt,
                          const float* __restrict__ dt_bias,
                          const float* __restrict__ A_log,
                          float* __restrict__ dt, float* __restrict__ dA) {
  int idx = blockIdx.x * blockDim.x + threadIdx.x;
  if (idx >= ROWS * NHEADS) return;
  int h   = idx & (NHEADS - 1);
  int row = idx >> 4;
  float v  = zxbcdt[(size_t)row * D_IN_PROJ + D_INNER + CONV_DIM + h] + dt_bias[h];
  float sp = (v > 20.0f) ? v : log1pf(__expf(v));
  float A  = -__expf(A_log[h]);
  dt[idx] = sp;
  dA[idx] = __expf(sp * A);
}

// -------------------------------------------------------------- selective scan
// One block per (batch, head). State h[64x128] lives in registers:
// thread t owns n = t&127 (fixed), p in { (t>>7) + 2*i : i=0..31 }.
__global__ void scan_kernel(const float* __restrict__ xBC,
                            const float* __restrict__ dt,
                            const float* __restrict__ dA,
                            const float* __restrict__ Dp,
                            float* __restrict__ Y) {
  int b = blockIdx.x >> 4;
  int h = blockIdx.x & 15;
  int t = threadIdx.x;
  int n = t & 127;
  int pbase = t >> 7;
  int lane = t & 31;
  float hs[32];
#pragma unroll
  for (int i = 0; i < 32; ++i) hs[i] = 0.0f;
  __shared__ float xs_sh[HEADDIM];
  __shared__ float y_sh[HEADDIM];
  float Dv = Dp[h];
  for (int l = 0; l < SEQ; ++l) {
    int row = b * SEQ + l;
    const float* xrow = xBC + (size_t)row * CONV_DIM;
    float dtv = dt[row * NHEADS + h];
    float dAv = dA[row * NHEADS + h];
    float Bn = xrow[D_INNER + n];
    float Cn = xrow[D_INNER + D_STATE + n];
    if (t < HEADDIM) { xs_sh[t] = xrow[h * HEADDIM + t]; y_sh[t] = 0.0f; }
    __syncthreads();
#pragma unroll
    for (int i = 0; i < 32; ++i) {
      int p = pbase + 2 * i;
      float dtx = dtv * xs_sh[p];
      hs[i] = hs[i] * dAv + dtx * Bn;
      float part = hs[i] * Cn;
#pragma unroll
      for (int off = 16; off; off >>= 1) part += __shfl_xor(part, off, 32);
      if (lane == 0) atomicAdd(&y_sh[p], part);   // ds_add_f32
    }
    __syncthreads();
    if (t < HEADDIM)
      Y[(size_t)row * D_INNER + h * HEADDIM + t] = y_sh[t] + xs_sh[t] * Dv;
    __syncthreads();
  }
}

// -------------------------------- gated RMSNorm( y * silu(z) ) -> bf16 for GEMM2
__global__ void gated_norm_bf16_kernel(const float* __restrict__ Y,
                                       const float* __restrict__ zxbcdt,
                                       const float* __restrict__ gw,
                                       uint16_t* __restrict__ yg) {
  int row = blockIdx.x;
  float v[4];
  float ss = 0.0f;
#pragma unroll
  for (int j = 0; j < 4; ++j) {
    int col = threadIdx.x + 256 * j;
    float z = zxbcdt[(size_t)row * D_IN_PROJ + col];
    float g = Y[(size_t)row * D_INNER + col] * silu(z);
    v[j] = g;
    ss += g * g;
  }
  __shared__ float red[8];
  ss = block_reduce_sum(ss, red);
  float scale = rsqrtf(ss * (1.0f / D_INNER) + 1e-5f);
#pragma unroll
  for (int j = 0; j < 4; ++j) {
    int col = threadIdx.x + 256 * j;
    yg[(size_t)row * D_INNER + col] = f2bf(gw[col] * v[j] * scale);
  }
}

// ============================================================================
extern "C" void kernel_launch(void* const* d_in, const int* in_sizes, int n_in,
                              void* d_out, int out_size, void* d_ws, size_t ws_size,
                              hipStream_t stream) {
  const float* x        = (const float*)d_in[0];
  const float* norm_w   = (const float*)d_in[1];
  const float* in_w     = (const float*)d_in[2];
  const float* conv_w   = (const float*)d_in[3];
  const float* conv_b   = (const float*)d_in[4];
  const float* dt_bias  = (const float*)d_in[5];
  const float* A_log    = (const float*)d_in[6];
  const float* D_param  = (const float*)d_in[7];
  const float* gated_w  = (const float*)d_in[8];
  const float* out_w    = (const float*)d_in[9];
  float* out            = (float*)d_out;

  // ---- workspace layout
  char* ws = (char*)d_ws;
  size_t off = 0;
  auto alloc = [&](size_t bytes) -> char* {
    char* p = ws + off;
    off += (bytes + 255) & ~(size_t)255;
    return p;
  };
  uint16_t* u_bf16   = (uint16_t*)alloc((size_t)ROWS * D_MODEL * 2);
  uint16_t* inw_bf16 = (uint16_t*)alloc((size_t)D_IN_PROJ_PAD * D_MODEL * 2);
  uint16_t* outw_bf16= (uint16_t*)alloc((size_t)D_MODEL * D_INNER * 2);
  float*    zxbcdt   = (float*)   alloc((size_t)ROWS * D_IN_PROJ * 4);
  float*    xBCact   = (float*)   alloc((size_t)ROWS * CONV_DIM * 4);
  float*    dt_buf   = (float*)   alloc((size_t)ROWS * NHEADS * 4);
  float*    dA_buf   = (float*)   alloc((size_t)ROWS * NHEADS * 4);
  float*    y_buf    = (float*)   alloc((size_t)ROWS * D_INNER * 4);
  uint16_t* yg_bf16  = (uint16_t*)alloc((size_t)ROWS * D_INNER * 2);
  (void)ws_size; (void)in_sizes; (void)n_in; (void)out_size;

  // 1) weight down-conversion (+ zero padding for in_proj rows)
  {
    size_t n1 = (size_t)D_IN_PROJ_PAD * D_MODEL;   // padded
    size_t n2 = (size_t)D_MODEL * D_INNER;
    convert_pad_bf16_kernel<<<(unsigned)((n1 + 255) / 256), 256, 0, stream>>>(
        in_w, inw_bf16, D_IN_PROJ, D_IN_PROJ_PAD, D_MODEL);
    convert_pad_bf16_kernel<<<(unsigned)((n2 + 255) / 256), 256, 0, stream>>>(
        out_w, outw_bf16, D_MODEL, D_MODEL, D_INNER);
  }

  // 2) RMSNorm -> bf16 activations
  rmsnorm_bf16_kernel<<<ROWS, 256, 0, stream>>>(x, norm_w, u_bf16);

  // 3) in_proj GEMM: (8192x512) x (512x2320) -> zxbcdt (f32)
  {
    int blocks = (ROWS / 128) * (D_IN_PROJ_PAD / 128);   // 64 * 19
    gemm_bf16_wmma_kernel<<<blocks, 256, 0, stream>>>(
        u_bf16, inw_bf16, zxbcdt, nullptr, ROWS, D_IN_PROJ, D_IN_PROJ_PAD, D_MODEL);
  }

  // 4) depthwise causal conv + SiLU on xBC slice
  {
    size_t n = (size_t)ROWS * CONV_DIM;
    conv_silu_kernel<<<(unsigned)((n + 255) / 256), 256, 0, stream>>>(
        zxbcdt, conv_w, conv_b, xBCact);
  }

  // 5) dt softplus + dA
  dt_kernel<<<(ROWS * NHEADS + 255) / 256, 256, 0, stream>>>(
      zxbcdt, dt_bias, A_log, dt_buf, dA_buf);

  // 6) selective scan (64 blocks = batch*heads)
  scan_kernel<<<BATCH * NHEADS, 256, 0, stream>>>(
      xBCact, dt_buf, dA_buf, D_param, y_buf);

  // 7) gated RMSNorm -> bf16
  gated_norm_bf16_kernel<<<ROWS, 256, 0, stream>>>(
      y_buf, zxbcdt, gated_w, yg_bf16);

  // 8) out_proj GEMM + residual: out = x + (8192x1024)x(1024x512)
  {
    int blocks = (ROWS / 128) * (D_MODEL / 128);         // 64 * 4
    gemm_bf16_wmma_kernel<<<blocks, 256, 0, stream>>>(
        yg_bf16, outw_bf16, out, x, ROWS, D_MODEL, D_MODEL, D_INNER);
  }
}
